// TemporalCF_5952824672319
// MI455X (gfx1250) — matrix operand order repaired and checked by reading, not compiled, not run on hardware
//
#include <hip/hip_runtime.h>
#include <hip/hip_bf16.h>

// TemporalCF fused forward for MI455X (gfx1250, wave32, WMMA).
//
// Roofline: ~4.3 GFLOP of GEMM vs ~25 MB of embedding-gather traffic
// (~1.1 us at 23.3 TB/s). fp32 gathers -> f16 LDS tiles ->
// v_wmma_f32_16x16x32_f16 (f32 accumulate) -> fp32 epilogue.
// Weights are pre-cast to f16 once (tcf_cvt_w): two global_load_b128 per B
// fragment, no f32->f16 VALU work in the GEMM loop. All four A fragments of a
// k-step are hoisted into distinct registers before the four WMMAs so the
// ds_load_b128s overlap (partial s_wait_dscnt) instead of serializing.

typedef __attribute__((ext_vector_type(16))) _Float16 v16h;
typedef __attribute__((ext_vector_type(8)))  _Float16 v8h;
typedef __attribute__((ext_vector_type(8)))  float    v8f;

#define BATCH     16384
#define DIM       128
#define TM        64          // batch rows per workgroup
#define NBINS     1000

#define W1_ELEMS  (256 * 384)
#define W2_ELEMS  (128 * 256)

// Workspace layout (bytes)
#define WS_MINMAX 0                       // 2 floats
#define WS_W1H    64                      // 256x384 f16 = 196608 B
#define WS_W2H    (WS_W1H + W1_ELEMS * 2) // 128x256 f16 =  65536 B
// total <= 262208 B

// LDS layout (halves padded by 8; rows stay 16B aligned)
#define XS_STRIDE 392         // 384 + 8
#define H1_STRIDE 264         // 256 + 8
#define H2_STRIDE 136         // 128 + 8
#define XS_OFF    0
#define H1_OFF    (TM * XS_STRIDE * 2)              // 50176
#define H2_OFF    (H1_OFF + TM * H1_STRIDE * 2)     // 83968
#define DEC_OFF   (H2_OFF + TM * H2_STRIDE * 2)     // 101376
#define UID_OFF   (DEC_OFF + TM * 4)
#define IID_OFF   (UID_OFF + TM * 4)
#define BIN_OFF   (IID_OFF + TM * 4)
#define SMEM_BYTES (BIN_OFF + TM * 4)               // 102400 B < 320 KB WGP LDS

// ---------------------------------------------------------------------------
// WMMA helpers (CDNA5 16x16x32 f16, wave32).
// A layout (this code's contiguous k order, 16-bit A 16x32): lane l: m = l&15, g = l>>4;
//   a[i] (i<8)  = A[m][kb + g*16 + i]
//   a[i] (i>=8) = A[m][kb + g*16 + 8 + (i-8)]
// B layout (K striped per lane-group): b[i] = B[kb + g*16 + i][n], n = l&15
// C/D layout: c[j] = D[mt*16 + g*8 + j][n]
// ---------------------------------------------------------------------------
__device__ __forceinline__ v8f wmma16(v16h a, v16h b, v8f c) {
    return __builtin_amdgcn_wmma_f32_16x16x32_f16(
        /*neg_a=*/false, a, /*neg_b=*/false, b,
        /*c_mod=*/(short)0, c, /*reuse_a=*/false, /*reuse_b=*/false);
}

__device__ __forceinline__ v16h load_a_lds(const _Float16* row, int kb, int g) {
    v8h lo = *(const v8h*)(row + kb + g * 16);        // ds_load_b128
    v8h hi = *(const v8h*)(row + kb + g * 16 + 8);   // ds_load_b128
    v16h a;
#pragma unroll
    for (int i = 0; i < 8; ++i) { a[i] = lo[i]; a[i + 8] = hi[i]; }
    return a;
}

// B fragment: 16 consecutive pre-cast f16 weights (32 B) -> two global_load_b128.
__device__ __forceinline__ v16h load_b_h(const _Float16* __restrict__ wrow, int k0) {
    return *(const v16h*)(wrow + k0);
}

// ---------------------------------------------------------------------------
// Kernel 0a: batch min/max of timestamps -> ws[0]=t_min, ws[1]=t_max
// ---------------------------------------------------------------------------
__global__ __launch_bounds__(256) void tcf_minmax(const int* __restrict__ ts,
                                                  float* __restrict__ ws) {
    __shared__ int smin[256];
    __shared__ int smax[256];
    const int tid = threadIdx.x;
    int mn = 0x7fffffff, mx = -2147483647 - 1;
    for (int i = tid; i < BATCH; i += 256) {
        int v = ts[i];
        mn = v < mn ? v : mn;
        mx = v > mx ? v : mx;
    }
    smin[tid] = mn; smax[tid] = mx;
    __syncthreads();
#pragma unroll
    for (int s = 128; s > 0; s >>= 1) {
        if (tid < s) {
            smin[tid] = smin[tid + s] < smin[tid] ? smin[tid + s] : smin[tid];
            smax[tid] = smax[tid + s] > smax[tid] ? smax[tid + s] : smax[tid];
        }
        __syncthreads();
    }
    if (tid == 0) { ws[0] = (float)smin[0]; ws[1] = (float)smax[0]; }
}

// ---------------------------------------------------------------------------
// Kernel 0b: one-shot f32 -> f16 pre-cast of W1 and W2 into workspace
// ---------------------------------------------------------------------------
__global__ __launch_bounds__(256) void tcf_cvt_w(const float* __restrict__ W1,
                                                 const float* __restrict__ W2,
                                                 _Float16* __restrict__ W1h,
                                                 _Float16* __restrict__ W2h) {
    const int i = blockIdx.x * 256 + threadIdx.x;
    if (i < W1_ELEMS) W1h[i] = (_Float16)W1[i];
    if (i < W2_ELEMS) W2h[i] = (_Float16)W2[i];
}

// ---------------------------------------------------------------------------
// Kernel 1: fused gather + decay + 3-layer MLP for a 64-row tile
// ---------------------------------------------------------------------------
__global__ __launch_bounds__(256) void tcf_main(
    const int*   __restrict__ user_ids,  const int*   __restrict__ item_ids,
    const int*   __restrict__ timestamps,
    const float* __restrict__ user_emb,  const float* __restrict__ item_emb,
    const float* __restrict__ time_emb,
    const float* __restrict__ user_bias, const float* __restrict__ item_bias,
    const float* __restrict__ time_decay,
    const _Float16* __restrict__ W1h, const float* __restrict__ b1,
    const _Float16* __restrict__ W2h, const float* __restrict__ b2,
    const float* __restrict__ W3, const float* __restrict__ b3,
    const float* __restrict__ minmax, float* __restrict__ out) {

    extern __shared__ char smem[];
    _Float16* xs   = (_Float16*)(smem + XS_OFF);
    _Float16* h1s  = (_Float16*)(smem + H1_OFF);
    _Float16* h2s  = (_Float16*)(smem + H2_OFF);
    float*    ldec = (float*)(smem + DEC_OFF);
    int*      luid = (int*)(smem + UID_OFF);
    int*      liid = (int*)(smem + IID_OFF);
    int*      lbin = (int*)(smem + BIN_OFF);

    const int tid  = threadIdx.x;
    const int r0   = blockIdx.x * TM;
    const int lane = tid & 31;
    const int wave = tid >> 5;
    const int nl   = lane & 15;
    const int g    = lane >> 4;

    const float tmin = minmax[0];
    const float tmax = minmax[1];

    // ---- Phase 0a: per-row scalars --------------------------------------
    if (tid < TM) {
        const int row = r0 + tid;
        const int u  = user_ids[row];
        const int it = item_ids[row];
        const float tsf = (float)timestamps[row];
        const float dec = expf(-time_decay[0] * (tsf - tmin));
        const float bin_size = fmaxf((tmax - tmin) / (float)NBINS, 1e-9f);
        int bin = (int)((tsf - tmin) / bin_size);
        bin = bin < 0 ? 0 : (bin > NBINS - 1 ? NBINS - 1 : bin);
        luid[tid] = u; liid[tid] = it; ldec[tid] = dec; lbin[tid] = bin;
    }
    __syncthreads();

    // ---- Phase 0b: gather x = [u*decay | v*decay | t] as f16 ------------
    for (int idx = tid; idx < TM * DIM; idx += 256) {
        const int row = idx >> 7;          // / 128
        const int col = idx & (DIM - 1);
        const float dec = ldec[row];
        const float uu = user_emb[(long)luid[row] * DIM + col] * dec;
        const float vv = item_emb[(long)liid[row] * DIM + col] * dec;
        const float tt = time_emb[lbin[row] * DIM + col];
        _Float16* xr = xs + row * XS_STRIDE;
        xr[col]           = (_Float16)uu;
        xr[DIM + col]     = (_Float16)vv;
        xr[2 * DIM + col] = (_Float16)tt;
    }
    __syncthreads();

    // ---- GEMM1: H1[64x256] = relu(X[64x384] @ W1^T + b1) ----------------
    // 16 N-tiles; each of 8 waves owns 2; all 4 M-tiles per wave.
#pragma unroll
    for (int t2 = 0; t2 < 2; ++t2) {
        const int nt  = wave * 2 + t2;
        const int col = nt * 16 + nl;
        const _Float16* wrow = W1h + col * 384;   // row n of W1 == col n of W1^T
        const float bias = b1[col];
        v8f acc0 = {}, acc1 = {}, acc2 = {}, acc3 = {};
#pragma unroll
        for (int ks = 0; ks < 12; ++ks) {
            const int kb = ks * 32;
            const v16h bf = load_b_h(wrow, kb + g * 16);
            // hoist all 4 A fragments -> 8 overlapping ds_load_b128
            const v16h a0 = load_a_lds(xs + (0 * 16 + nl) * XS_STRIDE, kb, g);
            const v16h a1 = load_a_lds(xs + (1 * 16 + nl) * XS_STRIDE, kb, g);
            const v16h a2 = load_a_lds(xs + (2 * 16 + nl) * XS_STRIDE, kb, g);
            const v16h a3 = load_a_lds(xs + (3 * 16 + nl) * XS_STRIDE, kb, g);
            acc0 = wmma16(a0, bf, acc0);
            acc1 = wmma16(a1, bf, acc1);
            acc2 = wmma16(a2, bf, acc2);
            acc3 = wmma16(a3, bf, acc3);
        }
        v8f accs[4] = {acc0, acc1, acc2, acc3};
#pragma unroll
        for (int mt = 0; mt < 4; ++mt) {
#pragma unroll
            for (int j = 0; j < 8; ++j) {
                const int m = mt * 16 + g * 8 + j;
                float h = accs[mt][j] + bias;
                h = h > 0.f ? h : 0.f;
                h1s[m * H1_STRIDE + col] = (_Float16)h;
            }
        }
    }
    __syncthreads();

    // ---- GEMM2: H2[64x128] = relu(H1[64x256] @ W2^T + b2) ---------------
    // 8 N-tiles; one per wave.
    {
        const int col = wave * 16 + nl;
        const _Float16* wrow = W2h + col * 256;
        const float bias = b2[col];
        v8f acc0 = {}, acc1 = {}, acc2 = {}, acc3 = {};
#pragma unroll
        for (int ks = 0; ks < 8; ++ks) {
            const int kb = ks * 32;
            const v16h bf = load_b_h(wrow, kb + g * 16);
            const v16h a0 = load_a_lds(h1s + (0 * 16 + nl) * H1_STRIDE, kb, g);
            const v16h a1 = load_a_lds(h1s + (1 * 16 + nl) * H1_STRIDE, kb, g);
            const v16h a2 = load_a_lds(h1s + (2 * 16 + nl) * H1_STRIDE, kb, g);
            const v16h a3 = load_a_lds(h1s + (3 * 16 + nl) * H1_STRIDE, kb, g);
            acc0 = wmma16(a0, bf, acc0);
            acc1 = wmma16(a1, bf, acc1);
            acc2 = wmma16(a2, bf, acc2);
            acc3 = wmma16(a3, bf, acc3);
        }
        v8f accs[4] = {acc0, acc1, acc2, acc3};
#pragma unroll
        for (int mt = 0; mt < 4; ++mt) {
#pragma unroll
            for (int j = 0; j < 8; ++j) {
                const int m = mt * 16 + g * 8 + j;
                float h = accs[mt][j] + bias;
                h = h > 0.f ? h : 0.f;
                h2s[m * H2_STRIDE + col] = (_Float16)h;
            }
        }
    }
    __syncthreads();

    // ---- Phase 3: out = H2 @ W3^T + b3 + user_bias + item_bias ----------
    // 4 lanes per row, 32 elements each, shfl-xor reduce within lane quad.
    {
        const int row = tid >> 2;
        const int q   = tid & 3;
        const _Float16* hrow = h2s + row * H2_STRIDE + q * 32;
        const float* w3q = W3 + q * 32;
        float sum = 0.f;
#pragma unroll
        for (int k = 0; k < 32; ++k) sum += (float)hrow[k] * w3q[k];
        sum += __shfl_xor(sum, 1, 32);
        sum += __shfl_xor(sum, 2, 32);
        if (q == 0) {
            out[r0 + row] = sum + b3[0] + user_bias[luid[row]] + item_bias[liid[row]];
        }
    }
}

// ---------------------------------------------------------------------------
extern "C" void kernel_launch(void* const* d_in, const int* in_sizes, int n_in,
                              void* d_out, int out_size, void* d_ws, size_t ws_size,
                              hipStream_t stream) {
    const int*   user_ids   = (const int*)d_in[0];
    const int*   item_ids   = (const int*)d_in[1];
    const int*   timestamps = (const int*)d_in[2];
    const float* user_emb   = (const float*)d_in[3];
    const float* item_emb   = (const float*)d_in[4];
    const float* time_emb   = (const float*)d_in[5];
    const float* user_bias  = (const float*)d_in[6];
    const float* item_bias  = (const float*)d_in[7];
    const float* time_decay = (const float*)d_in[8];
    const float* W1 = (const float*)d_in[9];
    const float* b1 = (const float*)d_in[10];
    const float* W2 = (const float*)d_in[11];
    const float* b2 = (const float*)d_in[12];
    const float* W3 = (const float*)d_in[13];
    const float* b3 = (const float*)d_in[14];

    char* ws = (char*)d_ws;
    float*     wsf  = (float*)(ws + WS_MINMAX);
    _Float16*  W1h  = (_Float16*)(ws + WS_W1H);
    _Float16*  W2h  = (_Float16*)(ws + WS_W2H);
    float*     outp = (float*)d_out;

    tcf_minmax<<<1, 256, 0, stream>>>(timestamps, wsf);
    tcf_cvt_w<<<(W1_ELEMS + 255) / 256, 256, 0, stream>>>(W1, W2, W1h, W2h);
    tcf_main<<<BATCH / TM, 256, SMEM_BYTES, stream>>>(
        user_ids, item_ids, timestamps,
        user_emb, item_emb, time_emb,
        user_bias, item_bias, time_decay,
        W1h, b1, W2h, b2, W3, b3,
        wsf, outp);
}
